// SS4Rec_15977278341431
// MI455X (gfx1250) — matrix operand order repaired and checked by reading, not compiled, not run on hardware
//
#include <hip/hip_runtime.h>
#include <hip/hip_bf16.h>
#include <math.h>

// ---------------------------------------------------------------------------
// SS4Rec forward for MI455X (gfx1250, wave32, WMMA + TDM).
//  - GEMMs: compile-time-specialized v_wmma_f32_16x16x32_bf16 tiles; A staged
//    to LDS with tensor_load_to_lds (TDM), B pre-packed to bf16 fragments.
//  - Producers emit bf16 activation copies once (halves GEMM read traffic).
//  - Scan / LN / head remain f32 VALU (latency-bound, non-matrix work).
// ---------------------------------------------------------------------------

typedef __attribute__((ext_vector_type(16))) __bf16 v16bf;
typedef __attribute__((ext_vector_type(8)))  __bf16 v8bf;
typedef __attribute__((ext_vector_type(8)))  float  v8f;
typedef unsigned int  uint32x4 __attribute__((ext_vector_type(4)));
typedef int           int32x4  __attribute__((ext_vector_type(4)));
typedef int           int32x8  __attribute__((ext_vector_type(8)));

#define BATCH   256
#define SEQL    200
#define DMODEL  128
#define DINNER  256
#define DSTATE  16
#define NROWS   (BATCH * SEQL)   // 51200

// ---- TDM availability (probe-verified builtin; arity differs per toolchain)
#if defined(__HIP_DEVICE_COMPILE__) && defined(__gfx1250__) && defined(__has_builtin)
#  if __has_builtin(__builtin_amdgcn_tensor_load_to_lds)
#    define SS4_HAVE_TDM 1
#  endif
#endif
#ifndef SS4_HAVE_TDM
#  define SS4_HAVE_TDM 0
#endif

__device__ __forceinline__ float silu_f(float x) {
    return x / (1.0f + __expf(-x));
}

// ---------------------------------------------------------------------------
// Stage one wave's 16 x K bf16 A-tile (row-major) into LDS.
// TDM: 2D descriptor, data_size=2B, tile 16 rows x K cols (ISA §8.3-8.5).
// Fallback: cooperative dword copy (per-wave LDS ops are in-order).
// ---------------------------------------------------------------------------
__device__ __forceinline__ void stage_a_tile(const __bf16* __restrict__ gsrc,
                                             __bf16* __restrict__ lds_dst,
                                             int K, int lane) {
#if SS4_HAVE_TDM
    unsigned long long ga = (unsigned long long)(uintptr_t)gsrc;
    unsigned lds_off = (unsigned)(uintptr_t)lds_dst;   // LDS address = low 32 bits of pointer

    uint32x4 g0;
    g0[0] = 1u;                                   // count=1, user descriptor
    g0[1] = lds_off;                              // lds_addr
    g0[2] = (unsigned)ga;                         // global_addr[31:0]
    g0[3] = (unsigned)(ga >> 32) | (2u << 30);    // global_addr[56:32] | type=2

    int32x8 g1;
    g1[0] = (1 << 16);                            // mask=0 | data_size=1 (2B)
    g1[1] = (K & 0xffff) << 16;                   // tensor_dim0[15:0]
    g1[2] = (16 << 16);                           // dim0 hi=0 | tensor_dim1=16
    g1[3] = (K & 0xffff) << 16;                   // dim1 hi=0 | tile_dim0=K
    g1[4] = 16;                                   // tile_dim1=16, tile_dim2=0
    g1[5] = K;                                    // tensor_dim0_stride[31:0]
    g1[6] = 0;
    g1[7] = 0;

    int32x4 g2 = {0, 0, 0, 0};
    int32x4 g3 = {0, 0, 0, 0};
#if __clang_major__ >= 23
    int32x8 g4 = {0, 0, 0, 0, 0, 0, 0, 0};
    __builtin_amdgcn_tensor_load_to_lds(g0, g1, g2, g3, g4, 0);
#else
    __builtin_amdgcn_tensor_load_to_lds(g0, g1, g2, g3, 0);
#endif
    __builtin_amdgcn_s_wait_tensorcnt(0);
    asm volatile("" ::: "memory");                // keep LDS reads below wait
    (void)lane;
#else
    const unsigned int* s = (const unsigned int*)gsrc;   // 2 bf16 per dword
    unsigned int* d = (unsigned int*)lds_dst;
    const int n = (16 * K) >> 1;
    for (int i = lane; i < n; i += 32) d[i] = s[i];
#endif
}

// ---------------------------------------------------------------------------
// Pack f32 weight W[K,N] (row-major) into bf16 WMMA B-fragment layout at
// destination tile column offset dst_nt0 (lets W_B / W_C share one buffer):
//   Wp[(((dst_nt0+nt)*(K/32) + kt)*32 + lane)*16 + e]
//     = W[(kt*32 + 8*(lane>>4) + e + (e>=8 ? 8 : 0))*N + nt*16 + (lane&15)]
// ---------------------------------------------------------------------------
__global__ void pack_weight_kernel(const float* __restrict__ W,
                                   __bf16* __restrict__ Wp,
                                   int N, int K, int dst_nt0) {
    int idx = blockIdx.x * blockDim.x + threadIdx.x;
    const int kt_total = K >> 5;
    const int total = (N >> 4) * kt_total * 32;
    if (idx >= total) return;
    const int lane = idx & 31;
    const int t = idx >> 5;
    const int kt = t % kt_total;
    const int nt = t / kt_total;
    const int g = lane >> 4;
    const int l15 = lane & 15;
    __bf16* dst = Wp + (((size_t)(dst_nt0 + nt) * kt_total + kt) * 32 + lane) * 16;
#pragma unroll
    for (int e = 0; e < 16; ++e) {
        int kk = kt * 32 + g * 8 + e + (e >= 8 ? 8 : 0);
        dst[e] = (__bf16)W[(size_t)kk * N + nt * 16 + l15];
    }
}

// ---------------------------------------------------------------------------
// Compile-time-specialized WMMA GEMM:  C[M,N] = Abf[M,K] @ W + bias.
//  - 128-thread block = 4 waves; wave w owns rows [blockIdx.y*64 + w*16, +16).
//  - Per K-step: 2 LDS b128 loads (A frag) then NSUB x (32B packed-bf16 B
//    fragment + v_wmma). Fully unrolled: no guards, no accumulator phi-copies.
//  - A fragment (16x32 bf16, ISA 7.12.2): elems 0-7 <- K run [k0+8g, +8),
//    elems 8-15 <- K run [k0+16+8g, +8), row = lane&15.
// ---------------------------------------------------------------------------
template <int N, int K, int NSUB>
__global__ void wmma_gemm_tpl(const __bf16* __restrict__ Abf,
                              const __bf16* __restrict__ Wp,
                              const float* __restrict__ bias,
                              float* __restrict__ C) {
    constexpr int KT = K / 32;
    __shared__ __bf16 tileA[4][16 * K];
    const int tid  = threadIdx.x;
    const int w    = tid >> 5;
    const int lane = tid & 31;
    const int g    = lane >> 4;
    const int l15  = lane & 15;
    const int mt   = blockIdx.y * 4 + w;       // this wave's 16-row tile
    const size_t rowBase = (size_t)mt * 16;
    __bf16* myTile = tileA[w];

    stage_a_tile(Abf + rowBase * K, myTile, K, lane);

    const int nt0 = blockIdx.x * NSUB;
    const __bf16* wpBase = Wp + ((size_t)nt0 * KT * 32 + lane) * 16;

    v8f acc[NSUB] = {};
#pragma unroll
    for (int kt = 0; kt < KT; ++kt) {
        const __bf16* arow = myTile + l15 * K + kt * 32;
        v8bf alo = *(const v8bf*)(arow + 8 * g);
        v8bf ahi = *(const v8bf*)(arow + 16 + 8 * g);
        v16bf afrag = __builtin_shufflevector(alo, ahi,
            0, 1, 2, 3, 4, 5, 6, 7, 8, 9, 10, 11, 12, 13, 14, 15);
#pragma unroll
        for (int i = 0; i < NSUB; ++i) {
            const v16bf bfrag = *(const v16bf*)(
                wpBase + ((size_t)i * KT + kt) * 32 * 16);
            acc[i] = __builtin_amdgcn_wmma_f32_16x16x32_bf16(
                false, afrag, false, bfrag, (short)0, acc[i], false, false);
        }
    }

#pragma unroll
    for (int i = 0; i < NSUB; ++i) {
        const int col = (nt0 + i) * 16 + l15;
        const float bv = bias ? bias[col] : 0.0f;
#pragma unroll
        for (int r = 0; r < 8; ++r) {
            C[(rowBase + g * 8 + r) * N + col] = acc[i][r] + bv;
        }
    }
}

// ---------------------------------------------------------------------------
// Embedding gather + tau; also emits bf16 activation copy for the first GEMM.
// ---------------------------------------------------------------------------
__global__ void embed_tau_kernel(const int* __restrict__ item_seq,
                                 const int* __restrict__ ts,
                                 const float* __restrict__ item_emb,
                                 const float* __restrict__ pos_emb,
                                 float* __restrict__ X,
                                 __bf16* __restrict__ Xbf,
                                 float* __restrict__ TAU) {
    size_t idx = (size_t)blockIdx.x * blockDim.x + threadIdx.x;
    if (idx >= (size_t)NROWS * DMODEL) return;
    size_t row = idx >> 7;
    int d = (int)(idx & 127);
    int l = (int)(row % SEQL);
    int item = item_seq[row];
    float v = item_emb[(size_t)item * DMODEL + d] + pos_emb[(size_t)l * DMODEL + d];
    X[idx] = v;
    Xbf[idx] = (__bf16)v;
    if (d == 0) {
        float tau;
        if (l == 0) {
            tau = 1.0f;
        } else {
            float diff = (float)(ts[row] - ts[row - 1]) / 86400.0f;
            if (diff == 0.0f) diff = 0.01f;
            diff = fminf(fmaxf(diff, 0.01f), 365.0f) / 365.0f;
            tau = fminf(fmaxf(diff, 1e-6f), 1.0f);
        }
        TAU[row] = tau;
    }
}

// ---------------------------------------------------------------------------
// Depthwise causal conv (K=4) + SiLU; writes f32 (for scan) and bf16 (GEMMs).
// ---------------------------------------------------------------------------
__global__ void conv_silu_kernel(const float* __restrict__ XZ,
                                 const float* __restrict__ cw,
                                 const float* __restrict__ cb,
                                 float* __restrict__ X1,
                                 __bf16* __restrict__ X1bf) {
    size_t idx = (size_t)blockIdx.x * blockDim.x + threadIdx.x;
    if (idx >= (size_t)NROWS * DINNER) return;
    size_t row = idx >> 8;
    int d = (int)(idx & 255);
    int l = (int)(row % SEQL);
    float acc = cb[d];
#pragma unroll
    for (int k = 0; k < 4; ++k) {
        int ls = l + k - 3;
        if (ls >= 0) {
            acc += XZ[(row + (size_t)(k - 3)) * (2 * DINNER) + d] * cw[d * 4 + k];
        }
    }
    float s = silu_f(acc);
    X1[idx] = s;
    X1bf[idx] = (__bf16)s;
}

// ---------------------------------------------------------------------------
// dt = clip(softplus(raw) * tau, 0.001, 0.1)   (in-place)
// ---------------------------------------------------------------------------
__global__ void dt_finalize_kernel(float* __restrict__ DTb,
                                   const float* __restrict__ TAU) {
    size_t idx = (size_t)blockIdx.x * blockDim.x + threadIdx.x;
    if (idx >= (size_t)NROWS * DINNER) return;
    size_t row = idx >> 8;
    float x = DTb[idx];
    float sp = (x > 20.0f) ? x : log1pf(__expf(x));
    float dt = sp * TAU[row];
    DTb[idx] = fminf(fmaxf(dt, 0.001f), 0.1f);
}

// ---------------------------------------------------------------------------
// Selective scan; combined B|C rows ([rows,32]: cols 0-15 = B, 16-31 = C)
// staged once in LDS (25.6 KB), barrier-free 200-step loop.
// Fuses the y * silu(z) gate and writes bf16 directly (only GEMM consumes it).
// ---------------------------------------------------------------------------
__global__ void ssm_scan_kernel(const float* __restrict__ X1,
                                const float* __restrict__ DTb,
                                const float* __restrict__ BCb,
                                const float* __restrict__ XZ,
                                const float* __restrict__ A_log,
                                const float* __restrict__ Dp,
                                __bf16* __restrict__ Ybf) {
    __shared__ float sBC[SEQL * 32];
    const int b = blockIdx.x;
    const int d = threadIdx.x;    // 0..255

    for (int i = threadIdx.x; i < SEQL * 32; i += blockDim.x) {
        sBC[i] = BCb[(size_t)b * SEQL * 32 + i];
    }
    __syncthreads();

    float a[DSTATE], h[DSTATE];
#pragma unroll
    for (int s = 0; s < DSTATE; ++s) {
        a[s] = -__expf(A_log[d * DSTATE + s]);
        h[s] = 0.0f;
    }
    const float dp = Dp[d];

    for (int t = 0; t < SEQL; ++t) {
        const size_t row = (size_t)b * SEQL + t;
        const float dt  = DTb[row * DINNER + d];
        const float xv  = X1[row * DINNER + d];
        const float dtx = dt * xv;
        const float* bt = &sBC[t * 32];
        float y = 0.0f;
#pragma unroll
        for (int s = 0; s < DSTATE; ++s) {
            float dA = __expf(dt * a[s]);
            h[s] = dA * h[s] + dtx * bt[s];
            y += h[s] * bt[16 + s];
        }
        y += xv * dp;
        const float z = XZ[row * (2 * DINNER) + DINNER + d];
        Ybf[row * DINNER + d] = (__bf16)(y * silu_f(z));
    }
}

// ---------------------------------------------------------------------------
// Row LayerNorm over D=128 (+optional residual, +optional bf16 side output).
// ---------------------------------------------------------------------------
__global__ void ln_kernel(const float* __restrict__ A,
                          const float* __restrict__ Res,    // may be nullptr
                          const float* __restrict__ g,
                          const float* __restrict__ bb,
                          float* __restrict__ out,
                          __bf16* __restrict__ obf) {       // may be nullptr
    __shared__ float sr[DMODEL];
    const size_t row = blockIdx.x;
    const int t = threadIdx.x;
    float v = A[row * DMODEL + t];
    if (Res) v += Res[row * DMODEL + t];
    sr[t] = v;
    __syncthreads();
    for (int s = 64; s > 0; s >>= 1) {
        if (t < s) sr[t] += sr[t + s];
        __syncthreads();
    }
    const float mu = sr[0] * (1.0f / DMODEL);
    __syncthreads();
    float dv = v - mu;
    sr[t] = dv * dv;
    __syncthreads();
    for (int s = 64; s > 0; s >>= 1) {
        if (t < s) sr[t] += sr[t + s];
        __syncthreads();
    }
    const float var = sr[0] * (1.0f / DMODEL);
    float r = dv * rsqrtf(var + 1e-5f) * g[t] + bb[t];
    out[row * DMODEL + t] = r;
    if (obf) obf[row * DMODEL + t] = (__bf16)r;
}

// ---------------------------------------------------------------------------
// Head MLP: one block per batch row (256->128->64->1).
// ---------------------------------------------------------------------------
__global__ void head_kernel(const float* __restrict__ X,
                            const int* __restrict__ users,
                            const int* __restrict__ targets,
                            const float* __restrict__ user_emb,
                            const float* __restrict__ item_emb,
                            const float* __restrict__ W1, const float* __restrict__ b1,
                            const float* __restrict__ W2, const float* __restrict__ b2,
                            const float* __restrict__ W3, const float* __restrict__ b3,
                            float* __restrict__ out) {
    __shared__ float h0[2 * DMODEL];
    __shared__ float h1[DMODEL];
    __shared__ float h2[DMODEL / 2];
    const int b = blockIdx.x;
    const int t = threadIdx.x;     // 128 threads
    const size_t lastRow = ((size_t)b * SEQL + (SEQL - 1)) * DMODEL;
    const int u  = users[b];
    const int it = targets[b];
    h0[t]          = user_emb[(size_t)u * DMODEL + t] + X[lastRow + t];
    h0[DMODEL + t] = item_emb[(size_t)it * DMODEL + t];
    __syncthreads();

    float acc = b1[t];
    for (int k = 0; k < 2 * DMODEL; ++k) acc += h0[k] * W1[k * DMODEL + t];
    h1[t] = fmaxf(acc, 0.0f);
    __syncthreads();

    if (t < DMODEL / 2) {
        float a2 = b2[t];
        for (int k = 0; k < DMODEL; ++k) a2 += h1[k] * W2[k * (DMODEL / 2) + t];
        h2[t] = fmaxf(a2, 0.0f);
    }
    __syncthreads();

    if (t == 0) {
        float a3 = b3[0];
        for (int k = 0; k < DMODEL / 2; ++k) a3 += h2[k] * W3[k];
        out[b] = a3;
    }
}

// ---------------------------------------------------------------------------
// Host orchestration.  d_in layout (setup_inputs order, params flattened):
//   0 users, 1 item_seq, 2 target_items, 3 timestamps,
//   4 item_emb, 5 user_emb, 6 pos_emb,
//   7+14*layer: {W_in,b_in,conv_w,conv_b,W_dt,b_dt,W_B,W_C,A_log,Dp,W_out,
//                b_out,ln_g,ln_b},  35 lnf_g, 36 lnf_b,
//   37 W1, 38 b1, 39 W2, 40 b2, 41 W3, 42 b3
// ---------------------------------------------------------------------------
extern "C" void kernel_launch(void* const* d_in, const int* in_sizes, int n_in,
                              void* d_out, int out_size, void* d_ws, size_t ws_size,
                              hipStream_t stream) {
    (void)in_sizes; (void)n_in; (void)out_size; (void)ws_size;

    const int*   users    = (const int*)d_in[0];
    const int*   item_seq = (const int*)d_in[1];
    const int*   targets  = (const int*)d_in[2];
    const int*   tstamps  = (const int*)d_in[3];
    const float* item_emb = (const float*)d_in[4];
    const float* user_emb = (const float*)d_in[5];
    const float* pos_emb  = (const float*)d_in[6];
    const float* lnf_g    = (const float*)d_in[35];
    const float* lnf_b    = (const float*)d_in[36];

    // Workspace carve-out (256B aligned)
    char* wsb = (char*)d_ws;
    size_t off = 0;
    auto alloc = [&](size_t bytes) -> char* {
        char* p = wsb + off;
        off += (bytes + 255) & ~(size_t)255;
        return p;
    };
    float*  X    = (float*) alloc((size_t)NROWS * DMODEL * 4);
    __bf16* Xbf  = (__bf16*)alloc((size_t)NROWS * DMODEL * 2);
    float*  TAU  = (float*) alloc((size_t)NROWS * 4);
    float*  XZ   = (float*) alloc((size_t)NROWS * 2 * DINNER * 4);
    float*  X1   = (float*) alloc((size_t)NROWS * DINNER * 4);
    __bf16* X1bf = (__bf16*)alloc((size_t)NROWS * DINNER * 2);
    float*  DTb  = (float*) alloc((size_t)NROWS * DINNER * 4);
    float*  BCb  = (float*) alloc((size_t)NROWS * 32 * 4);    // B|C combined
    __bf16* Ybf  = (__bf16*)alloc((size_t)NROWS * DINNER * 2);
    float*  OUTb = (float*) alloc((size_t)NROWS * DMODEL * 4);
    __bf16* Wp_in[2], *Wp_dt[2], *Wp_BC[2], *Wp_out[2];
    for (int l = 0; l < 2; ++l) {
        Wp_in[l]  = (__bf16*)alloc((size_t)DMODEL * 2 * DINNER * 2);
        Wp_dt[l]  = (__bf16*)alloc((size_t)DINNER * DINNER * 2);
        Wp_BC[l]  = (__bf16*)alloc((size_t)DINNER * 32 * 2);
        Wp_out[l] = (__bf16*)alloc((size_t)DINNER * DMODEL * 2);
    }

    auto pack = [&](const float* W, __bf16* Wp, int N, int K, int dst_nt0) {
        int total = (N >> 4) * (K >> 5) * 32;
        pack_weight_kernel<<<dim3((total + 127) / 128), dim3(128), 0, stream>>>(
            W, Wp, N, K, dst_nt0);
    };

    // 0) Pack all weights into WMMA fragment layout (tiny, L2-resident).
    for (int l = 0; l < 2; ++l) {
        const int p = 7 + l * 14;
        pack((const float*)d_in[p + 0],  Wp_in[l],  2 * DINNER, DMODEL, 0);
        pack((const float*)d_in[p + 4],  Wp_dt[l],  DINNER,     DINNER, 0);
        pack((const float*)d_in[p + 6],  Wp_BC[l],  DSTATE,     DINNER, 0); // W_B -> cols 0-15
        pack((const float*)d_in[p + 7],  Wp_BC[l],  DSTATE,     DINNER, 1); // W_C -> cols 16-31
        pack((const float*)d_in[p + 10], Wp_out[l], DMODEL,     DINNER, 0);
    }

    // 1) Embedding + tau
    {
        size_t total = (size_t)NROWS * DMODEL;
        embed_tau_kernel<<<dim3((unsigned)((total + 255) / 256)), dim3(256), 0, stream>>>(
            item_seq, tstamps, item_emb, pos_emb, X, Xbf, TAU);
    }

    // 2) Mamba blocks
    for (int layer = 0; layer < 2; ++layer) {
        const int p = 7 + layer * 14;
        const float* b_in  = (const float*)d_in[p + 1];
        const float* cw    = (const float*)d_in[p + 2];
        const float* cb    = (const float*)d_in[p + 3];
        const float* b_dt  = (const float*)d_in[p + 5];
        const float* A_log = (const float*)d_in[p + 8];
        const float* Dp    = (const float*)d_in[p + 9];
        const float* b_out = (const float*)d_in[p + 11];
        const float* ln_g  = (const float*)d_in[p + 12];
        const float* ln_b  = (const float*)d_in[p + 13];

        // xz = x @ W_in + b_in          (N=512, K=128, 8 subtiles/wave)
        wmma_gemm_tpl<512, 128, 8><<<dim3(4, NROWS / 64), dim3(128), 0, stream>>>(
            Xbf, Wp_in[layer], b_in, XZ);

        {
            size_t total = (size_t)NROWS * DINNER;
            conv_silu_kernel<<<dim3((unsigned)((total + 255) / 256)), dim3(256), 0, stream>>>(
                XZ, cw, cb, X1, X1bf);
        }

        // dt_raw = x1 @ W_dt + b_dt     (N=256, K=256)
        wmma_gemm_tpl<256, 256, 4><<<dim3(4, NROWS / 64), dim3(128), 0, stream>>>(
            X1bf, Wp_dt[layer], b_dt, DTb);
        {
            size_t total = (size_t)NROWS * DINNER;
            dt_finalize_kernel<<<dim3((unsigned)((total + 255) / 256)), dim3(256), 0, stream>>>(
                DTb, TAU);
        }

        // [Bm | Cm] = x1 @ [W_B | W_C]  (N=32, K=256, fused)
        wmma_gemm_tpl<32, 256, 2><<<dim3(1, NROWS / 64), dim3(128), 0, stream>>>(
            X1bf, Wp_BC[layer], nullptr, BCb);

        ssm_scan_kernel<<<dim3(BATCH), dim3(DINNER), 0, stream>>>(
            X1, DTb, BCb, XZ, A_log, Dp, Ybf);

        // out = y @ W_out + b_out       (N=128, K=256)
        wmma_gemm_tpl<128, 256, 4><<<dim3(2, NROWS / 64), dim3(128), 0, stream>>>(
            Ybf, Wp_out[layer], b_out, OUTb);

        // x = LN(res + out) ; bf16 copy feeds the next layer's in-proj GEMM
        ln_kernel<<<dim3(NROWS), dim3(DMODEL), 0, stream>>>(
            OUTb, X, ln_g, ln_b, X, (layer == 0) ? Xbf : nullptr);
    }

    // 3) Final LN (f32 only; head consumes f32)
    ln_kernel<<<dim3(NROWS), dim3(DMODEL), 0, stream>>>(
        X, nullptr, lnf_g, lnf_b, X, nullptr);

    // 4) Head MLP
    head_kernel<<<dim3(BATCH), dim3(DMODEL), 0, stream>>>(
        X, users, targets, user_emb, item_emb,
        (const float*)d_in[37], (const float*)d_in[38],
        (const float*)d_in[39], (const float*)d_in[40],
        (const float*)d_in[41], (const float*)d_in[42],
        (float*)d_out);
}
